// GATLayerExtractable_56736517980834
// MI455X (gfx1250) — matrix operand order repaired and checked by reading, not compiled, not run on hardware
//
#include <hip/hip_runtime.h>
#include <hip/hip_bf16.h>
#include <math.h>

// ---------------------------------------------------------------------------
// Problem constants (match reference)
#define BB   4
#define CC   2048
#define DDm  512
#define HHn  8
#define DKK  64          // DDm / HHn
#define EXPF 2
#define ROWS (BB*CC)     // 8192

typedef __bf16 bf16;
typedef __attribute__((ext_vector_type(16))) __bf16 v16bf;
typedef __attribute__((ext_vector_type(8)))  __bf16 bf16x8;
typedef __attribute__((ext_vector_type(8)))  float  v8f;

union Frag16 { v16bf v; bf16x8 h[2]; };

static __device__ __forceinline__ v8f zero8() {
    v8f z;
    #pragma unroll
    for (int i = 0; i < 8; ++i) z[i] = 0.0f;
    return z;
}

static __device__ __forceinline__ v8f wmma_bf16(v16bf a, v16bf b, v8f c) {
    return __builtin_amdgcn_wmma_f32_16x16x32_bf16(false, a, false, b,
                                                   (short)0, c, false, false);
}

// DPP lane permutation (pure VALU, no LDS round-trip).
template <int CTRL>
static __device__ __forceinline__ float dpp_mov(float x) {
    int t = __builtin_amdgcn_update_dpp(0, __builtin_bit_cast(int, x),
                                        CTRL, 0xf, 0xf, true);
    return __builtin_bit_cast(float, t);
}

// Max-reduce across each 16-lane group: xor1, xor2 (quad perms), then
// half-row mirror and row mirror. Result replicated across the 16 lanes.
static __device__ __forceinline__ float rowmax16(float x) {
    x = fmaxf(x, dpp_mov<0xB1>(x));   // quad_perm [1,0,3,2]
    x = fmaxf(x, dpp_mov<0x4E>(x));   // quad_perm [2,3,0,1]
    x = fmaxf(x, dpp_mov<0x141>(x));  // row_half_mirror
    x = fmaxf(x, dpp_mov<0x140>(x));  // row_mirror
    return x;
}

// ---------------------------------------------------------------------------
// Convert fp32 weights [K,N] row-major into WMMA B-fragment swizzled bf16:
// tile (kt,nt) covers K=[kt*32,+32), N=[nt*16,+16); within a tile, element
// (lane,i) holds W[kt*32 + 16*(lane/16) + i][nt*16 + lane%16]  (i contiguous).
__global__ void convert_swizzle_w(const float* __restrict__ W,
                                  bf16* __restrict__ Wsw, int K, int N) {
    int idx = blockIdx.x * 256 + threadIdx.x;
    if (idx >= K * N) return;
    int tile = idx >> 9, r = idx & 511;
    int ln = r >> 4, i = r & 15;
    int ntiles = N >> 4;
    int kt = tile / ntiles, nt = tile % ntiles;
    int k = kt * 32 + ((ln >> 4) << 4) + i;
    int n = nt * 16 + (ln & 15);
    Wsw[idx] = (bf16)W[(size_t)k * N + n];
}

// ---------------------------------------------------------------------------
// LayerNorm over D=512, output bf16. One block per row, 256 threads.
__global__ __launch_bounds__(256)
void ln_kernel(const float* __restrict__ x, const float* __restrict__ g,
               const float* __restrict__ bta, bf16* __restrict__ y) {
    const int t = threadIdx.x;
    const size_t base = (size_t)blockIdx.x * DDm;
    float v0 = x[base + t], v1 = x[base + 256 + t];
    float s = v0 + v1, sq = v0 * v0 + v1 * v1;
    #pragma unroll
    for (int mk = 1; mk < 32; mk <<= 1) {
        s  += __shfl_xor(s,  mk, 32);
        sq += __shfl_xor(sq, mk, 32);
    }
    __shared__ float rs[8], rq[8];
    const int w = t >> 5, lane = t & 31;
    if (lane == 0) { rs[w] = s; rq[w] = sq; }
    __syncthreads();
    s = 0.f; sq = 0.f;
    #pragma unroll
    for (int i = 0; i < 8; ++i) { s += rs[i]; sq += rq[i]; }
    const float mean = s * (1.0f / DDm);
    const float var  = sq * (1.0f / DDm) - mean * mean;
    const float rstd = rsqrtf(var + 1e-5f);
    y[base + t]       = (bf16)((v0 - mean) * rstd * g[t]       + bta[t]);
    y[base + 256 + t] = (bf16)((v1 - mean) * rstd * g[t + 256] + bta[t + 256]);
}

// ---------------------------------------------------------------------------
// Transpose V (from qkv buffer [row,1536], V at col 1024+h*64+d) into
// vt[b][h][d][c]  so P.V B-fragments are contiguous loads.
__global__ __launch_bounds__(256)
void transpose_v(const bf16* __restrict__ qkv, bf16* __restrict__ vt) {
    __shared__ bf16 tile[64][72];
    const int bh = blockIdx.x, c0 = blockIdx.y * 64;
    const int b = bh >> 3, h = bh & 7;
    #pragma unroll
    for (int it = 0; it < 16; ++it) {
        int idx = it * 256 + threadIdx.x;
        int kk = idx >> 6, dd = idx & 63;
        tile[kk][dd] = qkv[(size_t)(b * CC + c0 + kk) * (3 * DDm) + 2 * DDm + h * DKK + dd];
    }
    __syncthreads();
    #pragma unroll
    for (int it = 0; it < 16; ++it) {
        int idx = it * 256 + threadIdx.x;
        int dd = idx >> 6, kk = idx & 63;
        vt[((size_t)bh * DKK + dd) * CC + c0 + kk] = tile[kk][dd];
    }
}

// ---------------------------------------------------------------------------
// Generic bf16 WMMA GEMM: A[M,K] bf16 row-major, Bsw swizzled bf16, fp32 acc.
// 8 waves/block, each wave a 32x64 output tile -> block tile 256x64.
// MODE 0: out bf16 = acc+bias ; MODE 1: bf16 gelu(acc+bias) ;
// MODE 2: out f32 = acc+bias+resid.
template <int MODE>
__global__ __launch_bounds__(256)
void gemm_bf16(const bf16* __restrict__ A, const bf16* __restrict__ Bsw,
               const float* __restrict__ bias, const float* __restrict__ resid,
               void* __restrict__ outv, int M, int N, int K) {
    const int lane = threadIdx.x & 31, w = threadIdx.x >> 5;
    const int lmod = lane & 15, lhi = lane >> 4;
    const int m0 = blockIdx.x * 256 + w * 32;
    const int n0 = blockIdx.y * 64;
    const int ntiles = N >> 4;
    (void)M;

    v8f acc[2][4];
    #pragma unroll
    for (int i = 0; i < 2; ++i)
        #pragma unroll
        for (int j = 0; j < 4; ++j) acc[i][j] = zero8();

    for (int k0 = 0; k0 < K; k0 += 32) {
        Frag16 a[2];
        #pragma unroll
        for (int mf = 0; mf < 2; ++mf) {
            const size_t rb = (size_t)(m0 + mf * 16 + lmod) * K + k0 + lhi * 16;
            a[mf].h[0] = *(const bf16x8*)(A + rb);
            a[mf].h[1] = *(const bf16x8*)(A + rb + 8);
        }
        const size_t btile = (size_t)(k0 >> 5) * ntiles + (n0 >> 4);
        #pragma unroll
        for (int nf = 0; nf < 4; ++nf) {
            v16bf bfr = *(const v16bf*)(Bsw + (btile + nf) * 512 + (size_t)lane * 16);
            acc[0][nf] = wmma_bf16(a[0].v, bfr, acc[0][nf]);
            acc[1][nf] = wmma_bf16(a[1].v, bfr, acc[1][nf]);
        }
    }

    #pragma unroll
    for (int nf = 0; nf < 4; ++nf) {
        const int col = n0 + nf * 16 + lmod;
        const float bz = bias[col];
        #pragma unroll
        for (int mf = 0; mf < 2; ++mf) {
            #pragma unroll
            for (int r = 0; r < 8; ++r) {
                const int row = m0 + mf * 16 + r + 8 * lhi;
                float v = acc[mf][nf][r] + bz;
                const size_t oi = (size_t)row * N + col;
                if (MODE == 0) {
                    ((bf16*)outv)[oi] = (bf16)v;
                } else if (MODE == 1) {
                    v = 0.5f * v * (1.0f + erff(v * 0.70710678118654752f));
                    ((bf16*)outv)[oi] = (bf16)v;
                } else {
                    ((float*)outv)[oi] = v + resid[oi];
                }
            }
        }
    }
}

// ---------------------------------------------------------------------------
// Flash-style attention: each wave owns 16 q-rows x DK=64 for one (b,h).
// 8 waves/block -> 128 q rows/block. Blocks ordered head-fastest so the 8
// heads sharing one adj_bias tile are co-resident (L2 reuse of the 67MB
// bias stream). Online softmax over key tiles of 32.
//  - row-max reduction: DPP only (no LDS)
//  - row-sum: folded into the tensor path as a 5th WMMA accumulator (P * 1)
__global__ __launch_bounds__(256)
void attn_kernel(const bf16* __restrict__ qkv, const bf16* __restrict__ vt,
                 const float* __restrict__ adj, bf16* __restrict__ out) {
    const int lane = threadIdx.x & 31, w = threadIdx.x >> 5;
    const int lmod = lane & 15, lhi = lane >> 4;
    const int bh = blockIdx.x % (BB * HHn);
    const int qt = blockIdx.x / (BB * HHn);
    const int b = bh / HHn, h = bh % HHn;
    const int q0 = qt * 128 + w * 16;

    __shared__ __align__(32) bf16 pbuf[8][16 * 32];
    bf16* pw = &pbuf[w][0];

    // Preload Q fragments for both 32-wide d-tiles of DK=64
    Frag16 qf[2];
    {
        const size_t rowb = (size_t)(b * CC + q0 + lmod) * (3 * DDm);
        #pragma unroll
        for (int dh = 0; dh < 2; ++dh)
            #pragma unroll
            for (int c = 0; c < 2; ++c)
                qf[dh].h[c] = *(const bf16x8*)(qkv + rowb + h * DKK + dh * 32 + lhi * 16 + c * 8);
    }

    // ones B-fragment for the row-sum accumulator (P . 1)
    v16bf onesb;
    #pragma unroll
    for (int i = 0; i < 16; ++i) onesb[i] = (bf16)1.0f;

    v8f o[4];
    v8f lsum = zero8();     // row sums of P, replicated over lanes (C-layout)
    float m[8];
    #pragma unroll
    for (int i = 0; i < 4; ++i) o[i] = zero8();
    #pragma unroll
    for (int r = 0; r < 8; ++r) m[r] = -INFINITY;

    const float SCALE = 0.125f;               // DK^-0.5
    const float L2E   = 1.4426950408889634f;  // log2(e)
    const size_t adjrow0 = ((size_t)b * CC + q0) * CC;

    for (int kt = 0; kt < CC; kt += 32) {
        // S = Q . K^T   (K rows are contiguous in d -> direct B fragments)
        v8f s0 = zero8(), s1 = zero8();
        #pragma unroll
        for (int dh = 0; dh < 2; ++dh) {
            const bf16* kbase = qkv + (size_t)(b * CC + kt) * (3 * DDm)
                                + DDm + h * DKK + dh * 32 + lhi * 16;
            v16bf kb0 = *(const v16bf*)(kbase + (size_t)lmod * (3 * DDm));
            v16bf kb1 = *(const v16bf*)(kbase + (size_t)(16 + lmod) * (3 * DDm));
            s0 = wmma_bf16(qf[dh].v, kb0, s0);
            s1 = wmma_bf16(qf[dh].v, kb1, s1);
        }
        // fuse scale + adj_bias (fp32, coalesced across the 16-lane group)
        float sv[2][8];
        #pragma unroll
        for (int r = 0; r < 8; ++r) {
            const size_t arow = adjrow0 + (size_t)(r + 8 * lhi) * CC + kt;
            sv[0][r] = fmaf(s0[r], SCALE, adj[arow + lmod]);
            sv[1][r] = fmaf(s1[r], SCALE, adj[arow + 16 + lmod]);
        }
        // online softmax; row max via DPP, no cross-lane LDS traffic
        #pragma unroll
        for (int r = 0; r < 8; ++r) {
            const float mx = rowmax16(fmaxf(sv[0][r], sv[1][r]));
            const float mn = fmaxf(m[r], mx);
            const float alpha = exp2f((m[r] - mn) * L2E);
            m[r] = mn;
            sv[0][r] = exp2f((sv[0][r] - mn) * L2E);
            sv[1][r] = exp2f((sv[1][r] - mn) * L2E);
            lsum[r] *= alpha;
            #pragma unroll
            for (int dn = 0; dn < 4; ++dn) o[dn][r] *= alpha;
        }
        // C-layout P -> bf16 A-fragment via per-wave LDS bounce (no barrier:
        // same-wave LDS ops are in-order)
        #pragma unroll
        for (int f = 0; f < 2; ++f)
            #pragma unroll
            for (int r = 0; r < 8; ++r)
                pw[(r + 8 * lhi) * 32 + f * 16 + lmod] = (bf16)sv[f][r];
        asm volatile("s_wait_dscnt 0" ::: "memory");
        Frag16 pf;
        pf.h[0] = *(const bf16x8*)(pw + lmod * 32 + lhi * 16);
        pf.h[1] = *(const bf16x8*)(pw + lmod * 32 + lhi * 16 + 8);
        // O += P . V  and  lsum += P . 1  (row sums ride the WMMA pipe)
        const bf16* vb = vt + ((size_t)bh * DKK + lmod) * CC + kt + lhi * 16;
        #pragma unroll
        for (int dn = 0; dn < 4; ++dn) {
            v16bf vf = *(const v16bf*)(vb + (size_t)dn * 16 * CC);
            o[dn] = wmma_bf16(pf.v, vf, o[dn]);
        }
        lsum = wmma_bf16(pf.v, onesb, lsum);
    }

    // epilogue: O /= l, merge heads into [B,C,D] bf16
    #pragma unroll
    for (int r = 0; r < 8; ++r) {
        const float inv = 1.0f / lsum[r];
        const size_t orow = (size_t)(b * CC + q0 + r + 8 * lhi) * DDm + h * DKK;
        #pragma unroll
        for (int dn = 0; dn < 4; ++dn)
            out[orow + dn * 16 + lmod] = (bf16)(o[dn][r] * inv);
    }
}

// ---------------------------------------------------------------------------
extern "C" void kernel_launch(void* const* d_in, const int* in_sizes, int n_in,
                              void* d_out, int out_size, void* d_ws, size_t ws_size,
                              hipStream_t stream) {
    (void)in_sizes; (void)n_in; (void)out_size; (void)ws_size;
    const float* h     = (const float*)d_in[0];
    const float* adj   = (const float*)d_in[1];
    const float* w_qkv = (const float*)d_in[2];
    const float* b_qkv = (const float*)d_in[3];
    const float* w_out = (const float*)d_in[4];
    const float* b_out = (const float*)d_in[5];
    const float* ln1_g = (const float*)d_in[6];
    const float* ln1_b = (const float*)d_in[7];
    const float* ln2_g = (const float*)d_in[8];
    const float* ln2_b = (const float*)d_in[9];
    const float* w1    = (const float*)d_in[10];
    const float* b1    = (const float*)d_in[11];
    const float* w2    = (const float*)d_in[12];
    const float* b2    = (const float*)d_in[13];

    char* ws = (char*)d_ws;
    size_t off = 0;
    auto alloc = [&](size_t bytes) -> void* {
        void* p = ws + off;
        off = (off + bytes + 255) & ~(size_t)255;
        return p;
    };
    bf16*  hn   = (bf16*) alloc((size_t)ROWS * DDm * 2);
    bf16*  qkv  = (bf16*) alloc((size_t)ROWS * 3 * DDm * 2);
    bf16*  vtb  = (bf16*) alloc((size_t)BB * HHn * DKK * CC * 2);
    bf16*  aout = (bf16*) alloc((size_t)ROWS * DDm * 2);
    float* hmid = (float*)alloc((size_t)ROWS * DDm * 4);
    bf16*  hn2  = (bf16*) alloc((size_t)ROWS * DDm * 2);
    bf16*  ff1  = (bf16*) alloc((size_t)ROWS * DDm * EXPF * 2);
    bf16*  wqs  = (bf16*) alloc((size_t)DDm * 3 * DDm * 2);
    bf16*  wos  = (bf16*) alloc((size_t)DDm * DDm * 2);
    bf16*  w1s  = (bf16*) alloc((size_t)DDm * DDm * EXPF * 2);
    bf16*  w2s  = (bf16*) alloc((size_t)DDm * EXPF * DDm * 2);

    convert_swizzle_w<<<(DDm * 3 * DDm) / 256, 256, 0, stream>>>(w_qkv, wqs, DDm, 3 * DDm);
    convert_swizzle_w<<<(DDm * DDm) / 256,     256, 0, stream>>>(w_out, wos, DDm, DDm);
    convert_swizzle_w<<<(DDm * DDm * EXPF) / 256, 256, 0, stream>>>(w1, w1s, DDm, DDm * EXPF);
    convert_swizzle_w<<<(DDm * EXPF * DDm) / 256, 256, 0, stream>>>(w2, w2s, DDm * EXPF, DDm);

    ln_kernel<<<ROWS, 256, 0, stream>>>(h, ln1_g, ln1_b, hn);

    gemm_bf16<0><<<dim3(ROWS / 256, (3 * DDm) / 64), 256, 0, stream>>>(
        hn, wqs, b_qkv, nullptr, qkv, ROWS, 3 * DDm, DDm);

    transpose_v<<<dim3(BB * HHn, CC / 64), 256, 0, stream>>>(qkv, vtb);

    attn_kernel<<<(CC / 128) * BB * HHn, 256, 0, stream>>>(qkv, vtb, adj, aout);

    gemm_bf16<2><<<dim3(ROWS / 256, DDm / 64), 256, 0, stream>>>(
        aout, wos, b_out, h, hmid, ROWS, DDm, DDm);

    ln_kernel<<<ROWS, 256, 0, stream>>>(hmid, ln2_g, ln2_b, hn2);

    gemm_bf16<1><<<dim3(ROWS / 256, (DDm * EXPF) / 64), 256, 0, stream>>>(
        hn2, w1s, b1, nullptr, ff1, ROWS, DDm * EXPF, DDm);

    gemm_bf16<2><<<dim3(ROWS / 256, DDm / 64), 256, 0, stream>>>(
        ff1, w2s, b2, hmid, (float*)d_out, ROWS, DDm, DDm * EXPF);
}
